// Model_54812372632328
// MI455X (gfx1250) — matrix operand order repaired and checked by reading, not compiled, not run on hardware
//
#include <hip/hip_runtime.h>
#include <hip/hip_bf16.h>
#include <math.h>

// ---------------------------------------------------------------------------
// Model dims
// ---------------------------------------------------------------------------
#define BB   64
#define LL   512
#define MM   7
#define BM   448          // B*M
#define PATCH 16
#define STRIDE 8
#define PNUM 64           // patch_num
#define DD   768
#define NH   12
#define HD   64
#define NLAYER 12
#define DFF  3072
#define PLEN 4
#define POOL 30
#define PRED 96
#define TT   68           // PLEN + PNUM
#define TP   80           // T padded to 5x16
#define KP   96           // K padded for P@V (3x32)
#define NROW (BM*TT)      // 30464 transformer rows
#define EROW (BM*PNUM)    // 28672 emb rows
#define KIN  64           // padded patch K (48 -> 64)
#define HEADM (BM*3)      // 1344
#define HEADK (TT*DD/3)   // 17408
#define OUTN 43008        // 64*96*7

typedef __attribute__((ext_vector_type(16))) _Float16 v16h;
typedef __attribute__((ext_vector_type(8)))  _Float16 v8h;
typedef __attribute__((ext_vector_type(8)))  float    v8f;
typedef __attribute__((ext_vector_type(4)))  int      v4i;

__device__ __forceinline__ v16h mk16(v8h lo, v8h hi) {
  return __builtin_shufflevector(lo, hi, 0,1,2,3,4,5,6,7,8,9,10,11,12,13,14,15);
}

// ---------------------------------------------------------------------------
// gfx1250 async global->LDS copy (ASYNCcnt path), guarded so we fall back to
// load + ds_store staging if this toolchain doesn't expose the builtins.
// Probe-derived signature: (v4i AS1* src, v4i AS3* dst, imm offset, imm cpol).
// ---------------------------------------------------------------------------
#if defined(__has_builtin)
#if __has_builtin(__builtin_amdgcn_global_load_async_to_lds_b128) && \
    __has_builtin(__builtin_amdgcn_s_wait_asynccnt)
#define HAVE_ASYNC 1
#endif
#endif
#ifndef HAVE_ASYNC
#define HAVE_ASYNC 0
#endif

__device__ __forceinline__ void cp16_lds(void* lds, const void* g) {
#if HAVE_ASYNC
  __builtin_amdgcn_global_load_async_to_lds_b128(
      (__attribute__((address_space(1))) v4i*)g,
      (__attribute__((address_space(3))) v4i*)lds, 0, 0);
#else
  *(v8h*)lds = *(const v8h*)g;
#endif
}
__device__ __forceinline__ void async_wait0() {
#if HAVE_ASYNC
  __builtin_amdgcn_s_wait_asynccnt(0);
#endif
}

// db4 filters (decomposition low/high); rec filters are their reverses.
__constant__ float c_dlo[8] = {
  -0.010597401784997278f,  0.032883011666982945f,  0.030841381835986965f,
  -0.18703481171888114f,  -0.02798376941698385f,   0.6308807679295904f,
   0.7148465705525415f,    0.23037781330885523f };
__constant__ float c_dhi[8] = {
  -0.23037781330885523f,   0.7148465705525415f,   -0.6308807679295904f,
  -0.02798376941698385f,   0.18703481171888114f,   0.030841381835986965f,
  -0.032883011666982945f, -0.010597401784997278f };

// ---------------------------------------------------------------------------
// 1) per-(b,m) mean/std over L, write normalized transposed rows (BM x L)
// ---------------------------------------------------------------------------
__global__ __launch_bounds__(256) void stats_kernel(
    const float* __restrict__ x, float* __restrict__ meanv,
    float* __restrict__ stdv, float* __restrict__ xrows)
{
  int bm = blockIdx.x;
  int b = bm / MM, m = bm % MM;
  int tid = threadIdx.x;
  __shared__ float red[256];
  __shared__ float s_mu, s_sd;
  float s = 0.f, s2 = 0.f;
  for (int l = tid; l < LL; l += 256) {
    float v = x[((size_t)b * LL + l) * MM + m];
    s += v; s2 += v * v;
  }
  red[tid] = s; __syncthreads();
  for (int o = 128; o > 0; o >>= 1) { if (tid < o) red[tid] += red[tid + o]; __syncthreads(); }
  float mu = red[0] / LL; __syncthreads();
  red[tid] = s2; __syncthreads();
  for (int o = 128; o > 0; o >>= 1) { if (tid < o) red[tid] += red[tid + o]; __syncthreads(); }
  if (tid == 0) {
    float var = red[0] / LL - mu * mu;
    float sd = sqrtf(var + 1e-5f);
    meanv[bm] = mu; stdv[bm] = sd; s_mu = mu; s_sd = sd;
  }
  __syncthreads();
  float mu2 = s_mu, inv = 1.f / s_sd;
  for (int l = tid; l < LL; l += 256) {
    float v = x[((size_t)b * LL + l) * MM + m];
    xrows[(size_t)bm * LL + l] = (v - mu2) * inv;
  }
}

// ---------------------------------------------------------------------------
// 2) wavelet decomposition per row (exact symmetric-pad DWT/IDWT chain)
// ---------------------------------------------------------------------------
__device__ __forceinline__ float symext(const float* x, int n, int q) {
  int qq = q - 7;
  if (qq < 0) qq = -1 - qq;
  else if (qq >= n) qq = 2 * n - 1 - qq;
  return x[qq];
}
__device__ void dwt_step(const float* in, int n, const float* f, float* out,
                         int nout, int tid, int nt) {
  for (int i = tid; i < nout; i += nt) {
    float acc = 0.f;
#pragma unroll
    for (int j = 0; j < 8; ++j) acc += symext(in, n, 2 * i + j + 1) * f[7 - j];
    out[i] = acc;
  }
}
__device__ void idwt_step(const float* a, const float* d, int m, float* out,
                          int tid, int nt) {
  int nout = 2 * m - 6;
  for (int r = tid; r < nout; r += nt) {
    float acc = 0.f;
#pragma unroll
    for (int j = 0; j < 8; ++j) {
      int q = r + j - 1;
      if (q >= 0 && q < 2 * m - 1 && ((q & 1) == 0)) {
        int ui = q >> 1;
        if (a) acc += a[ui] * c_dlo[j];
        if (d) acc += d[ui] * c_dhi[j];
      }
    }
    out[r] = acc;
  }
}
__global__ __launch_bounds__(128) void wavelet_kernel(
    const float* __restrict__ xrows, float* __restrict__ dec)
{
  __shared__ float sx[512], sa[264], sb[136], sc[72], sd[72], t1[520], t2[520];
  int bm = blockIdx.x, tid = threadIdx.x;
  const float* xr = xrows + (size_t)bm * LL;
  for (int i = tid; i < 512; i += 128) sx[i] = xr[i];
  __syncthreads();
  dwt_step(sx, 512, c_dlo, sa, 259, tid, 128); __syncthreads();
  dwt_step(sa, 259, c_dlo, sb, 133, tid, 128); __syncthreads();
  dwt_step(sb, 133, c_dlo, sc, 70, tid, 128);
  dwt_step(sb, 133, c_dhi, sd, 70, tid, 128); __syncthreads();
  // trend: a3 -> 134 -> 262 -> 518
  idwt_step(sc, nullptr, 70, t1, tid, 128);  __syncthreads();
  idwt_step(t1, nullptr, 134, t2, tid, 128); __syncthreads();
  idwt_step(t2, nullptr, 262, t1, tid, 128); __syncthreads();
  // seasonal: d3 -> 134 -> 262 -> 518
  idwt_step(nullptr, sd, 70, t2, tid, 128);  __syncthreads();
  idwt_step(t2, nullptr, 134, sa, tid, 128); __syncthreads();
  idwt_step(sa, nullptr, 262, t2, tid, 128); __syncthreads();
  float* base = dec + (size_t)bm * 3 * 520;
  for (int i = tid; i < 520; i += 128) {
    int l = (i < 512) ? i : 511;   // edge pad
    float tr = t1[l], se = t2[l], re = sx[l] - tr - se;
    base[i] = tr; base[520 + i] = se; base[1040 + i] = re;
  }
}

// ---------------------------------------------------------------------------
// 3) patch extraction -> f16 (K padded 48 -> 64)
// ---------------------------------------------------------------------------
__global__ void patches_kernel(const float* __restrict__ dec,
                               _Float16* __restrict__ p16)
{
  size_t i = (size_t)blockIdx.x * 256 + threadIdx.x;   // EROW * KIN
  if (i >= (size_t)EROW * KIN) return;
  int c = (int)(i & 63);
  size_t r = i >> 6;
  int p = (int)(r & 63);
  size_t bm = r >> 6;
  float v = 0.f;
  if (c < 48) {
    int comp = c >> 4, off = c & 15;
    v = dec[(bm * 3 + comp) * 520 + p * STRIDE + off];
  }
  p16[i] = (_Float16)v;
}

// ---------------------------------------------------------------------------
// Weight f32 -> f16 transposed (N-major: wt[n*K+k] = w[k*N+n]); layer via grid.y
// ---------------------------------------------------------------------------
__global__ void convert_wt(const float* __restrict__ w, _Float16* __restrict__ wt,
                           int K, int N)
{
  size_t layer = blockIdx.y;
  const float* wl = w + layer * (size_t)K * N;
  _Float16* wtl = wt + layer * (size_t)K * N;
  size_t total = (size_t)K * N;
  for (size_t i = (size_t)blockIdx.x * 256 + threadIdx.x; i < total;
       i += (size_t)gridDim.x * 256) {
    size_t n = i / K, k = i % K;
    wtl[i] = (_Float16)wl[k * (size_t)N + n];
  }
}
__global__ void convert_inw(const float* __restrict__ w, _Float16* __restrict__ wt)
{
  int i = blockIdx.x * 256 + threadIdx.x;   // 768*64
  if (i >= DD * KIN) return;
  int n = i >> 6, k = i & 63;
  wt[i] = (_Float16)((k < 48) ? w[k * DD + n] : 0.f);
}

// ---------------------------------------------------------------------------
// Generic WMMA GEMM: C(MxN) = A(MxK,f16 row-major) @ Bt(NxK,f16)^T
// 8 waves/block, each wave: 16 x (16*NT) tile; v_wmma_f32_16x16x32_f16.
// B tile (NT*16 x 32) is staged in LDS, double-buffered: the next K-step's
// tile is copied global->LDS (async on gfx1250, ASYNCcnt) while WMMAs consume
// the current one. A fragments stream from global (2 x b128 per lane) with a
// prefetch hint ahead of the stream.
// MODE 0: Cf = acc+bias           MODE 1: emb->h row remap, Cf = acc+bias
// MODE 2: Cf += acc+bias          MODE 3: Ch = f16(acc+bias)
// MODE 4: Ch = f16(gelu_tanh(acc+bias))
// ---------------------------------------------------------------------------
template<int NT, int MODE>
__global__ __launch_bounds__(256) void gemm_wmma(
    const _Float16* __restrict__ A, const _Float16* __restrict__ Bt,
    const float* __restrict__ bias, float* __restrict__ Cf,
    _Float16* __restrict__ Ch, int M, int N, int K, int ldc)
{
  __shared__ __align__(16) _Float16 sB[2][NT * 16 * 32];
  const int tid = threadIdx.x;
  const int wave = tid >> 5;
  const int lane = tid & 31;
  const int mtile = blockIdx.x * 8 + wave;
  const bool mvalid = (mtile * 16 < M);
  const int ntile0 = blockIdx.y * NT;
  const int lm = lane & 15;
  const int hi = lane >> 4;

  // staging map: thread -> (row, 8-half chunk) of the NT*16 x 32 B tile
  const int srow = tid >> 2;
  const int schunk = (tid & 3) * 8;
  const bool sact = (srow < NT * 16);
  const _Float16* bsrc = Bt + (size_t)(ntile0 * 16 + srow) * K + schunk;

  v8f acc[NT];
#pragma unroll
  for (int t = 0; t < NT; ++t)
#pragma unroll
    for (int i = 0; i < 8; ++i) acc[t][i] = 0.0f;

  const int arow = mvalid ? (mtile * 16 + lm) : lm;   // clamp for idle waves
  const _Float16* aptr = A + (size_t)arow * K + (hi ? 8 : 0);

  // prologue: stage first B tile
  if (sact) cp16_lds(&sB[0][srow * 32 + schunk], bsrc);
  async_wait0();
  __syncthreads();

  int buf = 0;
  for (int kk = 0; kk < K; kk += 32) {
    if (kk + 32 < K && sact)
      cp16_lds(&sB[buf ^ 1][srow * 32 + schunk], bsrc + kk + 32);
    if (kk + 64 < K)
      __builtin_prefetch(aptr + kk + 64, 0, 1);   // global_prefetch_b8

    v8h a0 = *(const v8h*)(aptr + kk);
    v8h a1 = *(const v8h*)(aptr + kk + 16);
    v16h af = mk16(a0, a1);
#pragma unroll
    for (int t = 0; t < NT; ++t) {
      const _Float16* bp = &sB[buf][(t * 16 + lm) * 32 + (hi ? 8 : 0)];
      v8h b0 = *(const v8h*)bp;
      v8h b1 = *(const v8h*)(bp + 16);
      v16h bf = mk16(b0, b1);
      acc[t] = __builtin_amdgcn_wmma_f32_16x16x32_f16(
          false, af, false, bf, (short)0, acc[t], false, false);
    }
    async_wait0();
    __syncthreads();
    buf ^= 1;
  }

  if (mvalid) {
#pragma unroll
    for (int t = 0; t < NT; ++t) {
      const int col = (ntile0 + t) * 16 + lm;
      if (col >= N) continue;
      const float bv = bias ? bias[col] : 0.0f;
#pragma unroll
      for (int v = 0; v < 8; ++v) {
        const int row = mtile * 16 + v + hi * 8;
        if (row >= M) continue;
        float val = acc[t][v] + bv;
        if (MODE == 0) {
          Cf[(size_t)row * ldc + col] = val;
        } else if (MODE == 1) {
          int bm = row >> 6, p = row & 63;
          Cf[((size_t)bm * TT + PLEN + p) * ldc + col] = val;
        } else if (MODE == 2) {
          Cf[(size_t)row * ldc + col] += val;
        } else if (MODE == 3) {
          Ch[(size_t)row * ldc + col] = (_Float16)val;
        } else {
          float x3 = val * val * val;
          float g = 0.5f * val *
                    (1.0f + tanhf(0.7978845608028654f * (val + 0.044715f * x3)));
          Ch[(size_t)row * ldc + col] = (_Float16)g;
        }
      }
    }
  }
}

// ---------------------------------------------------------------------------
// LayerNorm row kernel: f32 in -> f16 out
// ---------------------------------------------------------------------------
__global__ __launch_bounds__(256) void ln_kernel(
    const float* __restrict__ x, const float* __restrict__ g,
    const float* __restrict__ b, _Float16* __restrict__ out, int ncols)
{
  int row = blockIdx.x, tid = threadIdx.x;
  const float* xr = x + (size_t)row * ncols;
  __shared__ float red[256];
  __shared__ float s_mu, s_rs;
  float s = 0.f, s2 = 0.f;
  for (int i = tid; i < ncols; i += 256) { float v = xr[i]; s += v; s2 += v * v; }
  red[tid] = s; __syncthreads();
  for (int o = 128; o > 0; o >>= 1) { if (tid < o) red[tid] += red[tid + o]; __syncthreads(); }
  float mu = red[0] / ncols; __syncthreads();
  red[tid] = s2; __syncthreads();
  for (int o = 128; o > 0; o >>= 1) { if (tid < o) red[tid] += red[tid + o]; __syncthreads(); }
  if (tid == 0) { s_mu = mu; s_rs = rsqrtf(red[0] / ncols - mu * mu + 1e-5f); }
  __syncthreads();
  float mu2 = s_mu, rs = s_rs;
  for (int i = tid; i < ncols; i += 256)
    out[(size_t)row * ncols + i] = (_Float16)((xr[i] - mu2) * rs * g[i] + b[i]);
}

// ---------------------------------------------------------------------------
// Fused attention per (seq, head): S = qk^T (WMMA), masked softmax, O = P V (WMMA)
// 5 waves (160 threads). Static LDS 58368B with region aliasing (P over q/k).
// q/k land in LDS via async copies (gfx1250 ASYNCcnt path) when available.
// ---------------------------------------------------------------------------
__global__ __launch_bounds__(160) void attn_kernel(
    const _Float16* __restrict__ qkv, _Float16* __restrict__ out)
{
  __shared__ __align__(16) char smem[58368];
  _Float16* sq  = (_Float16*)smem;                 // [80][64]  10240 B
  _Float16* sk  = sq + TP * HD;                    // [80][64]  10240 B
  _Float16* svT = (_Float16*)(smem + 20480);       // [64][96]  12288 B
  float*    sS  = (float*)(smem + 32768);          // [80][80]  25600 B
  _Float16* sP  = (_Float16*)smem;                 // [80][96]  aliases sq/sk

  const int s = blockIdx.x, hgrp = blockIdx.y;
  const int tid = threadIdx.x;
  const size_t base = (size_t)s * TT * (3 * DD) + (size_t)hgrp * HD;

  // q, k rows [0,68) straight global->LDS copies (async on gfx1250)
  for (int idx = tid; idx < TT * 8; idx += 160) {
    int t = idx >> 3, c = (idx & 7) * 8;
    const _Float16* g = qkv + base + (size_t)t * (3 * DD) + c;
    cp16_lds(sq + t * HD + c, g);
    cp16_lds(sk + t * HD + c, g + DD);
  }
  // zero pad rows [68,80)
  for (int idx = tid; idx < (TP - TT) * 8; idx += 160) {
    int t = TT + (idx >> 3), c = (idx & 7) * 8;
    v8h z; for (int j = 0; j < 8; ++j) z[j] = (_Float16)0.f;
    *(v8h*)(sq + t * HD + c) = z;
    *(v8h*)(sk + t * HD + c) = z;
  }
  // v transposed (svT[d][t]), rows t in [0,96): manual scatter
  for (int idx = tid; idx < KP * 8; idx += 160) {
    int t = idx >> 3, c = (idx & 7) * 8;
    v8h v; for (int j = 0; j < 8; ++j) v[j] = (_Float16)0.f;
    if (t < TT) v = *(const v8h*)(qkv + base + (size_t)t * (3 * DD) + 2 * DD + c);
    for (int j = 0; j < 8; ++j) svT[(c + j) * KP + t] = v[j];
  }
  async_wait0();
  __syncthreads();

  const int wave = tid >> 5, lane = tid & 31, lm = lane & 15, hi = lane >> 4;
  // S = q @ k^T  (each wave: one 16-row tile x 5 col tiles, K=64)
  for (int ct = 0; ct < 5; ++ct) {
    v8f acc;
#pragma unroll
    for (int i = 0; i < 8; ++i) acc[i] = 0.f;
#pragma unroll
    for (int kk = 0; kk < HD; kk += 32) {
      const _Float16* ap = sq + (wave * 16 + lm) * HD + kk + (hi ? 8 : 0);
      v16h af = mk16(*(const v8h*)ap, *(const v8h*)(ap + 16));
      const _Float16* bp = sk + (ct * 16 + lm) * HD + kk + (hi ? 8 : 0);
      v16h bf = mk16(*(const v8h*)bp, *(const v8h*)(bp + 16));
      acc = __builtin_amdgcn_wmma_f32_16x16x32_f16(
          false, af, false, bf, (short)0, acc, false, false);
    }
#pragma unroll
    for (int v = 0; v < 8; ++v)
      sS[(wave * 16 + v + hi * 8) * TP + ct * 16 + lm] = acc[v] * 0.125f;
  }
  __syncthreads();

  // causal softmax rows -> sP (f16, cols padded to 96)
  for (int r = tid; r < TP; r += 160) {
    if (r >= TT) {
      for (int j = 0; j < KP; ++j) sP[r * KP + j] = (_Float16)0.f;
      continue;
    }
    float mx = -1e30f;
    for (int j = 0; j <= r; ++j) mx = fmaxf(mx, sS[r * TP + j]);
    float sum = 0.f;
    for (int j = 0; j <= r; ++j) {
      float e = expf(sS[r * TP + j] - mx);
      sS[r * TP + j] = e; sum += e;
    }
    float inv = 1.f / sum;
    for (int j = 0; j < KP; ++j)
      sP[r * KP + j] = (_Float16)((j <= r) ? sS[r * TP + j] * inv : 0.f);
  }
  __syncthreads();

  // O = P @ V  (each wave: 16-row tile x 4 col tiles, K=96)
  for (int ct = 0; ct < 4; ++ct) {
    v8f acc;
#pragma unroll
    for (int i = 0; i < 8; ++i) acc[i] = 0.f;
#pragma unroll
    for (int kk = 0; kk < KP; kk += 32) {
      const _Float16* ap = sP + (wave * 16 + lm) * KP + kk + (hi ? 8 : 0);
      v16h af = mk16(*(const v8h*)ap, *(const v8h*)(ap + 16));
      const _Float16* bp = svT + (ct * 16 + lm) * KP + kk + (hi ? 8 : 0);
      v16h bf = mk16(*(const v8h*)bp, *(const v8h*)(bp + 16));
      acc = __builtin_amdgcn_wmma_f32_16x16x32_f16(
          false, af, false, bf, (short)0, acc, false, false);
    }
#pragma unroll
    for (int v = 0; v < 8; ++v) {
      int t = wave * 16 + v + hi * 8;
      if (t < TT)
        out[((size_t)s * TT + t) * DD + hgrp * HD + ct * 16 + lm] = (_Float16)acc[v];
    }
  }
}

// ---------------------------------------------------------------------------
// Prompt pool: xm = mean(emb), cosine sim top-k, build h = [prompt | emb] + wpe
// ---------------------------------------------------------------------------
__global__ void xm_kernel(const float* __restrict__ h, float* __restrict__ xm)
{
  int i = blockIdx.x * 256 + threadIdx.x;   // BM * D
  if (i >= BM * DD) return;
  int d = i % DD, bm = i / DD;
  float s = 0.f;
  for (int p = 0; p < PNUM; ++p)
    s += h[((size_t)bm * TT + PLEN + p) * DD + d];
  xm[i] = s * (1.f / PNUM);
}
__global__ void topk_kernel(const float* __restrict__ xm,
                            const float* __restrict__ pkey,
                            int* __restrict__ topi, float* __restrict__ simred)
{
  int bm = blockIdx.x * blockDim.x + threadIdx.x;
  if (bm >= BM) return;
  const float* q = xm + (size_t)bm * DD;
  float qn = 0.f;
  for (int d = 0; d < DD; ++d) qn += q[d] * q[d];
  qn = sqrtf(qn) + 1e-12f;
  float sim[POOL];
  for (int i = 0; i < POOL; ++i) {
    const float* k = pkey + (size_t)i * DD;
    float dot = 0.f, kn = 0.f;
    for (int d = 0; d < DD; ++d) { dot += q[d] * k[d]; kn += k[d] * k[d]; }
    sim[i] = dot / (qn * (sqrtf(kn) + 1e-12f));
  }
  bool used[POOL];
  for (int i = 0; i < POOL; ++i) used[i] = false;
  float ssum = 0.f;
  for (int j = 0; j < PLEN; ++j) {
    int bi = 0; float bv = -1e30f;
    for (int i = 0; i < POOL; ++i)
      if (!used[i] && sim[i] > bv) { bv = sim[i]; bi = i; }
    used[bi] = true;
    topi[bm * PLEN + j] = bi;
    ssum += bv;
  }
  simred[bm] = ssum;
}
__global__ __launch_bounds__(256) void reduce_sim_kernel(
    const float* __restrict__ simred, float* __restrict__ out)
{
  __shared__ float red[256];
  float s = 0.f;
  for (int i = threadIdx.x; i < BM; i += 256) s += simred[i];
  red[threadIdx.x] = s; __syncthreads();
  for (int o = 128; o > 0; o >>= 1) {
    if (threadIdx.x < o) red[threadIdx.x] += red[threadIdx.x + o];
    __syncthreads();
  }
  if (threadIdx.x == 0) out[0] = red[0] / (float)BM;
}
__global__ void prompt_wpe_kernel(float* __restrict__ h,
                                  const float* __restrict__ prompt,
                                  const float* __restrict__ wpe,
                                  const int* __restrict__ topi)
{
  size_t i = (size_t)blockIdx.x * 256 + threadIdx.x;   // NROW * D
  if (i >= (size_t)NROW * DD) return;
  int d = (int)(i % DD);
  size_t r = i / DD;
  int t = (int)(r % TT);
  size_t bm = r / TT;
  float w = wpe[(size_t)t * DD + d];
  if (t < PLEN)
    h[i] = prompt[(size_t)topi[bm * PLEN + t] * DD + d] + w;
  else
    h[i] += w;
}

// ---------------------------------------------------------------------------
// Head epilogue: sum 3 components, de-normalize, transpose to (B, PRED, M)
// ---------------------------------------------------------------------------
__global__ void final_out_kernel(const float* __restrict__ flat,
                                 const float* __restrict__ meanv,
                                 const float* __restrict__ stdv,
                                 float* __restrict__ out)
{
  int i = blockIdx.x * 256 + threadIdx.x;   // 64*96*7
  if (i >= OUTN) return;
  int m = i % MM;
  int t = (i / MM) % PRED;
  int b = i / (PRED * MM);
  int bm = b * MM + m;
  size_t rr = (size_t)bm * 3;
  float s = flat[rr * PRED + t] + flat[(rr + 1) * PRED + t] + flat[(rr + 2) * PRED + t];
  out[i] = s * stdv[bm] + meanv[bm];
}

// ---------------------------------------------------------------------------
// Host orchestration
// ---------------------------------------------------------------------------
extern "C" void kernel_launch(void* const* d_in, const int* in_sizes, int n_in,
                              void* d_out, int out_size, void* d_ws, size_t ws_size,
                              hipStream_t stream)
{
  (void)in_sizes; (void)n_in; (void)out_size; (void)ws_size;
  const float* x_enc      = (const float*)d_in[0];
  const float* in_w       = (const float*)d_in[4];
  const float* in_b       = (const float*)d_in[5];
  const float* prompt     = (const float*)d_in[6];
  const float* prompt_key = (const float*)d_in[7];
  const float* wpe        = (const float*)d_in[8];
  const float* ln1_g      = (const float*)d_in[9];
  const float* ln1_b      = (const float*)d_in[10];
  const float* qkv_w      = (const float*)d_in[11];
  const float* qkv_b      = (const float*)d_in[12];
  const float* proj_w     = (const float*)d_in[13];
  const float* proj_b     = (const float*)d_in[14];
  const float* ln2_g      = (const float*)d_in[15];
  const float* ln2_b      = (const float*)d_in[16];
  const float* fc_w       = (const float*)d_in[17];
  const float* fc_b       = (const float*)d_in[18];
  const float* mp_w       = (const float*)d_in[19];
  const float* mp_b       = (const float*)d_in[20];
  const float* lnf_g      = (const float*)d_in[21];
  const float* lnf_b      = (const float*)d_in[22];
  const float* out_w      = (const float*)d_in[23];
  const float* out_b      = (const float*)d_in[24];
  float* outp = (float*)d_out;

  char* ws = (char*)d_ws;
  size_t off = 0;
  auto alloc = [&](size_t bytes) -> char* {
    char* p = ws + off;
    off = (off + bytes + 255) & ~(size_t)255;
    return p;
  };
  float*     meanv    = (float*)alloc(BM * 4);
  float*     stdv     = (float*)alloc(BM * 4);
  float*     xrows    = (float*)alloc((size_t)BM * LL * 4);
  float*     dec      = (float*)alloc((size_t)BM * 3 * 520 * 4);
  _Float16*  p16      = (_Float16*)alloc((size_t)EROW * KIN * 2);
  float*     h        = (float*)alloc((size_t)NROW * DD * 4);
  _Float16*  a16      = (_Float16*)alloc((size_t)NROW * DD * 2);
  _Float16*  qkv16    = (_Float16*)alloc((size_t)NROW * 3 * DD * 2);
  _Float16*  attno16  = (_Float16*)alloc((size_t)NROW * DD * 2);
  _Float16*  mlp16    = (_Float16*)alloc((size_t)NROW * DFF * 2);
  float*     xm       = (float*)alloc((size_t)BM * DD * 4);
  int*       topi     = (int*)alloc(BM * PLEN * 4);
  float*     simred   = (float*)alloc(BM * 4);
  float*     flatout  = (float*)alloc((size_t)HEADM * PRED * 4);
  _Float16*  inwt     = (_Float16*)alloc((size_t)DD * KIN * 2);
  _Float16*  qkvwt    = (_Float16*)alloc((size_t)NLAYER * DD * 3 * DD * 2);
  _Float16*  projwt   = (_Float16*)alloc((size_t)NLAYER * DD * DD * 2);
  _Float16*  fcwt     = (_Float16*)alloc((size_t)NLAYER * DD * DFF * 2);
  _Float16*  mpwt     = (_Float16*)alloc((size_t)NLAYER * DFF * DD * 2);
  _Float16*  outwt    = (_Float16*)alloc((size_t)PRED * HEADK * 2);

  // weight conversion (transpose to N-major f16)
  convert_wt<<<dim3(2048, NLAYER), 256, 0, stream>>>(qkv_w, qkvwt, DD, 3 * DD);
  convert_wt<<<dim3(1024, NLAYER), 256, 0, stream>>>(proj_w, projwt, DD, DD);
  convert_wt<<<dim3(2048, NLAYER), 256, 0, stream>>>(fc_w, fcwt, DD, DFF);
  convert_wt<<<dim3(2048, NLAYER), 256, 0, stream>>>(mp_w, mpwt, DFF, DD);
  convert_wt<<<dim3(2048, 1), 256, 0, stream>>>(out_w, outwt, HEADK, PRED);
  convert_inw<<<(DD * KIN + 255) / 256, 256, 0, stream>>>(in_w, inwt);

  // front-end
  stats_kernel<<<BM, 256, 0, stream>>>(x_enc, meanv, stdv, xrows);
  wavelet_kernel<<<BM, 128, 0, stream>>>(xrows, dec);
  patches_kernel<<<(int)(((size_t)EROW * KIN + 255) / 256), 256, 0, stream>>>(dec, p16);

  // emb = patches @ in_w + in_b  -> h rows [bm*68+4+p]  (MODE 1)
  gemm_wmma<4, 1><<<dim3(EROW / 128, DD / 64), 256, 0, stream>>>(
      p16, inwt, in_b, h, nullptr, EROW, DD, KIN, DD);

  // prompt pool
  xm_kernel<<<(BM * DD + 255) / 256, 256, 0, stream>>>(h, xm);
  topk_kernel<<<2, 256, 0, stream>>>(xm, prompt_key, topi, simred);
  reduce_sim_kernel<<<1, 256, 0, stream>>>(simred, outp + OUTN);
  prompt_wpe_kernel<<<(int)(((size_t)NROW * DD + 255) / 256), 256, 0, stream>>>(
      h, prompt, wpe, topi);

  // transformer layers
  for (int i = 0; i < NLAYER; ++i) {
    ln_kernel<<<NROW, 256, 0, stream>>>(h, ln1_g + i * DD, ln1_b + i * DD, a16, DD);
    gemm_wmma<4, 3><<<dim3(NROW / 128, (3 * DD) / 64), 256, 0, stream>>>(
        a16, qkvwt + (size_t)i * DD * 3 * DD, qkv_b + (size_t)i * 3 * DD,
        nullptr, qkv16, NROW, 3 * DD, DD, 3 * DD);
    attn_kernel<<<dim3(BM, NH), 160, 0, stream>>>(qkv16, attno16);
    gemm_wmma<4, 2><<<dim3(NROW / 128, DD / 64), 256, 0, stream>>>(
        attno16, projwt + (size_t)i * DD * DD, proj_b + (size_t)i * DD,
        h, nullptr, NROW, DD, DD, DD);
    ln_kernel<<<NROW, 256, 0, stream>>>(h, ln2_g + i * DD, ln2_b + i * DD, a16, DD);
    gemm_wmma<4, 4><<<dim3(NROW / 128, DFF / 64), 256, 0, stream>>>(
        a16, fcwt + (size_t)i * DD * DFF, fc_b + (size_t)i * DFF,
        nullptr, mlp16, NROW, DFF, DD, DFF);
    gemm_wmma<4, 2><<<dim3(NROW / 128, DD / 64), 256, 0, stream>>>(
        mlp16, mpwt + (size_t)i * DFF * DD, mp_b + (size_t)i * DD,
        h, nullptr, NROW, DD, DFF, DD);
  }

  // final LN (f16 out doubles as the flattened (1344 x 17408) head input)
  ln_kernel<<<NROW, 256, 0, stream>>>(h, lnf_g, lnf_b, a16, DD);
  gemm_wmma<2, 0><<<dim3((HEADM + 127) / 128, PRED / 32), 256, 0, stream>>>(
      a16, outwt, out_b, flatout, nullptr, HEADM, PRED, HEADK, PRED);
  final_out_kernel<<<(OUTN + 255) / 256, 256, 0, stream>>>(flatout, meanv, stdv, outp);
}
